// FractalBlock_14259291422728
// MI455X (gfx1250) — matrix operand order repaired and checked by reading, not compiled, not run on hardware
//
#include <hip/hip_runtime.h>
#include <hip/hip_bf16.h>
#include <math.h>

typedef __bf16 bf16;
typedef __attribute__((ext_vector_type(16))) __bf16 v16bf;
typedef __attribute__((ext_vector_type(8)))  float  v8f;
typedef unsigned int u32x4 __attribute__((ext_vector_type(4)));
typedef int          i32x4 __attribute__((ext_vector_type(4)));
typedef int          i32x8 __attribute__((ext_vector_type(8)));

#define B_  2
#define T_  2048
#define D_  2048
#define H_  16
#define HD_ 128
#define FF_ 4096
#define M_  (B_*T_)   // 4096 token rows

#if __has_builtin(__builtin_amdgcn_global_load_async_to_lds_b128)
#define HAVE_ASYNC_LDS 1
#else
#define HAVE_ASYNC_LDS 0
#endif
#if __has_builtin(__builtin_amdgcn_tensor_load_to_lds)
#define HAVE_TDM 1
#else
#define HAVE_TDM 0
#endif

union FragBF { uint4 u[2]; v16bf v; };
union U4BF   { uint4 u; bf16 e[8]; };

static __device__ __forceinline__ v8f wmma_bf16(v16bf a, v16bf b, v8f c) {
  // D = A(16x32 bf16) * B(32x16 bf16) + C(16x16 f32)
  return __builtin_amdgcn_wmma_f32_16x16x32_bf16(false, a, false, b, (short)0, c, false, false);
}

// 16B global -> LDS, async (ASYNCcnt) when available, VGPR staging otherwise.
static __device__ __forceinline__ void async_copy16(const bf16* g, bf16* l) {
#if HAVE_ASYNC_LDS
  __builtin_amdgcn_global_load_async_to_lds_b128(
      (__attribute__((address_space(1))) i32x4*)g,
      (__attribute__((address_space(3))) i32x4*)l, 0, 0);
#else
  *(uint4*)l = *(const uint4*)g;
#endif
}
static __device__ __forceinline__ void async_wait() {
#if HAVE_ASYNC_LDS
#if __has_builtin(__builtin_amdgcn_s_wait_asynccnt)
  __builtin_amdgcn_s_wait_asynccnt(0);
#else
  asm volatile("s_wait_asynccnt 0x0" ::: "memory");
#endif
#endif
}

// Tensor Data Mover: 2D tile load (tile_x contiguous elems of 2B, tile_rows rows,
// row stride in elements).  D# packing per CDNA5 ISA.
static __device__ __forceinline__ void tdm_load_tile_2d(
    const void* gsrc, unsigned lds_byte_off, unsigned tile_x, unsigned tile_rows,
    unsigned long long row_stride_elems) {
#if HAVE_TDM
  const unsigned long long ga = (unsigned long long)(size_t)gsrc;
  u32x4 g0 = {0, 0, 0, 0};
  i32x8 g1 = {0, 0, 0, 0, 0, 0, 0, 0};
  i32x4 g2 = {0, 0, 0, 0};
  i32x4 g3 = {0, 0, 0, 0};
  g0[0] = 1u;                                                   // count=1 (valid user D#)
  g0[1] = lds_byte_off;                                         // lds_addr
  g0[2] = (unsigned)ga;                                         // global_addr[31:0]
  g0[3] = (unsigned)((ga >> 32) & 0x01FFFFFFull) | (2u << 30);  // global_addr[56:32], type=2
  g1[0] = (int)(1u << 16);                                      // wg_mask=0, data_size=1 (2B)
  g1[1] = (int)((tile_x & 0xffffu) << 16);                      // tensor_dim0[15:0]
  g1[2] = (int)(((tile_x >> 16) & 0xffffu) | ((tile_rows & 0xffffu) << 16)); // dim0[31:16], dim1[15:0]
  g1[3] = (int)(((tile_rows >> 16) & 0xffffu) | ((tile_x & 0xffffu) << 16)); // dim1[31:16], tile_dim0
  g1[4] = (int)(tile_rows & 0xffffu);                           // tile_dim1, tile_dim2=0
  g1[5] = (int)(unsigned)(row_stride_elems & 0xffffffffull);    // tensor_dim0_stride[31:0]
  g1[6] = (int)(unsigned)((row_stride_elems >> 32) & 0xffffull);// stride[47:32], dim1_stride=0
#if __clang_major__ >= 23
  i32x8 g4 = {0, 0, 0, 0, 0, 0, 0, 0};
  __builtin_amdgcn_tensor_load_to_lds(g0, g1, g2, g3, g4, 0);
#else
  __builtin_amdgcn_tensor_load_to_lds(g0, g1, g2, g3, 0);
#endif
#else
  (void)gsrc; (void)lds_byte_off; (void)tile_x; (void)tile_rows; (void)row_stride_elems;
#endif
}
static __device__ __forceinline__ void tdm_wait() {
#if HAVE_TDM && __has_builtin(__builtin_amdgcn_s_wait_tensorcnt)
  __builtin_amdgcn_s_wait_tensorcnt(0);
#endif
}

// ---------------------------------------------------------------------------
// fp32 -> bf16 weight conversion (grid-stride)
// ---------------------------------------------------------------------------
__global__ void cvt_f32_bf16(const float* __restrict__ s, bf16* __restrict__ d, size_t n) {
  size_t i = (size_t)blockIdx.x * blockDim.x + threadIdx.x;
  size_t stride = (size_t)gridDim.x * blockDim.x;
  for (; i < n; i += stride) d[i] = (bf16)s[i];
}

// ---------------------------------------------------------------------------
// RMSNorm + scale/shift -> bf16.  One row (D=2048) per 256-thread block.
// ---------------------------------------------------------------------------
__global__ __launch_bounds__(256) void rmsnorm_affine(
    const float* __restrict__ x, const float* __restrict__ nw,
    const float* __restrict__ gamma, const float* __restrict__ beta,
    bf16* __restrict__ out) {
  const int row = blockIdx.x;
  const float* xr = x + (size_t)row * D_;
  bf16* orow = out + (size_t)row * D_;
  __shared__ float red[8];
  const int tid = threadIdx.x, wave = tid >> 5, lane = tid & 31;
  float ss = 0.f;
  for (int i = tid; i < D_; i += 256) { float v = xr[i]; ss += v * v; }
#pragma unroll
  for (int off = 16; off >= 1; off >>= 1) ss += __shfl_xor(ss, off);
  if (lane == 0) red[wave] = ss;
  __syncthreads();
  if (tid == 0) { float t = 0.f; for (int i = 0; i < 8; ++i) t += red[i]; red[0] = t; }
  __syncthreads();
  const float r = rsqrtf(red[0] * (1.0f / D_) + 1.1920928955078125e-07f);
  for (int i = tid; i < D_; i += 256)
    orow[i] = (bf16)(xr[i] * r * nw[i] * gamma[i] + beta[i]);
}

// ---------------------------------------------------------------------------
// C[M,N] = A[M,K] @ W[N,K]^T  (both bf16, K-contiguous), f32 accumulation.
// Block tile 128x128, K-step 32.  Tiles staged with the Tensor Data Mover
// (one 2D descriptor per tile, wave 0 issues, TENSORcnt wait) when available.
// EPI==0: store bf16; EPI==1: store f32 with residual add.
// ---------------------------------------------------------------------------
template <int EPI>
__global__ __launch_bounds__(256) void gemm_bf16_tn(
    const bf16* __restrict__ A, const bf16* __restrict__ W,
    void* __restrict__ Cv, const float* __restrict__ res,
    int M, int N, int K) {
  __shared__ __align__(16) bf16 As[128][32];
  __shared__ __align__(16) bf16 Ws[128][32];
  const int m0 = blockIdx.y * 128, n0 = blockIdx.x * 128;
  const int tid = threadIdx.x, wave = tid >> 5, lane = tid & 31;
  const int half = lane >> 4, l16 = lane & 15;
#if HAVE_TDM
  const unsigned ldsA = (unsigned)(size_t)(void*)&As[0][0];   // generic LDS addr low 32b = LDS offset
  const unsigned ldsW = (unsigned)(size_t)(void*)&Ws[0][0];
#endif
  v8f acc[8] = {};
  for (int k0 = 0; k0 < K; k0 += 32) {
#if HAVE_TDM
    if (wave == 0) {
      tdm_load_tile_2d(&A[(size_t)m0 * K + k0], ldsA, 32u, 128u, (unsigned long long)K);
      tdm_load_tile_2d(&W[(size_t)n0 * K + k0], ldsW, 32u, 128u, (unsigned long long)K);
      tdm_wait();
    }
#else
    for (int c = tid; c < 512; c += 256) {
      const int row = c >> 2, col = (c & 3) << 3;
      async_copy16(&A[(size_t)(m0 + row) * K + (k0 + col)], &As[row][col]);
      async_copy16(&W[(size_t)(n0 + row) * K + (k0 + col)], &Ws[row][col]);
    }
    async_wait();
#endif
    if (k0 + 32 < K) {  // global_prefetch_b8 for the next K tile
      __builtin_prefetch(&A[(size_t)(m0 + (tid >> 1)) * K + (k0 + 32)], 0, 1);
      __builtin_prefetch(&W[(size_t)(n0 + (tid >> 1)) * K + (k0 + 32)], 0, 1);
    }
    __syncthreads();
    FragBF af, bw[8];  // batch all fragment loads, then back-to-back WMMAs
    af.u[0] = *(const uint4*)&As[wave * 16 + l16][half * 8];
    af.u[1] = *(const uint4*)&As[wave * 16 + l16][half * 8 + 16];
#pragma unroll
    for (int nt = 0; nt < 8; ++nt) {
      bw[nt].u[0] = *(const uint4*)&Ws[nt * 16 + l16][half * 8];
      bw[nt].u[1] = *(const uint4*)&Ws[nt * 16 + l16][half * 8 + 16];
    }
#pragma unroll
    for (int nt = 0; nt < 8; ++nt) acc[nt] = wmma_bf16(af.v, bw[nt].v, acc[nt]);
    __syncthreads();
  }
#pragma unroll
  for (int nt = 0; nt < 8; ++nt) {
#pragma unroll
    for (int r = 0; r < 8; ++r) {
      const int gm = m0 + wave * 16 + half * 8 + r;
      const int gn = n0 + nt * 16 + l16;
      const size_t idx = (size_t)gm * N + gn;
      if (EPI == 0) ((bf16*)Cv)[idx] = (bf16)acc[nt][r];
      else          ((float*)Cv)[idx] = res[idx] + acc[nt][r];
    }
  }
}

// ---------------------------------------------------------------------------
// Fused gate/up GEMM + SiLU*mul epilogue -> bf16 ff (async-LDS staged).
// ---------------------------------------------------------------------------
__global__ __launch_bounds__(256) void gemm_gateup(
    const bf16* __restrict__ A, const bf16* __restrict__ Wg, const bf16* __restrict__ Wu,
    bf16* __restrict__ FFo, int M, int N, int K) {
  __shared__ __align__(16) bf16 As[128][32];
  __shared__ __align__(16) bf16 Gs[128][32];
  __shared__ __align__(16) bf16 Us[128][32];
  const int m0 = blockIdx.y * 128, n0 = blockIdx.x * 128;
  const int tid = threadIdx.x, wave = tid >> 5, lane = tid & 31;
  const int half = lane >> 4, l16 = lane & 15;
  v8f accg[8] = {}, accu[8] = {};
  for (int k0 = 0; k0 < K; k0 += 32) {
    for (int c = tid; c < 512; c += 256) {
      const int row = c >> 2, col = (c & 3) << 3;
      async_copy16(&A[(size_t)(m0 + row) * K + (k0 + col)], &As[row][col]);
      async_copy16(&Wg[(size_t)(n0 + row) * K + (k0 + col)], &Gs[row][col]);
      async_copy16(&Wu[(size_t)(n0 + row) * K + (k0 + col)], &Us[row][col]);
    }
    async_wait();
    __syncthreads();
    FragBF af;
    af.u[0] = *(const uint4*)&As[wave * 16 + l16][half * 8];
    af.u[1] = *(const uint4*)&As[wave * 16 + l16][half * 8 + 16];
#pragma unroll
    for (int nt = 0; nt < 8; ++nt) {
      FragBF bg, bu;
      bg.u[0] = *(const uint4*)&Gs[nt * 16 + l16][half * 8];
      bg.u[1] = *(const uint4*)&Gs[nt * 16 + l16][half * 8 + 16];
      bu.u[0] = *(const uint4*)&Us[nt * 16 + l16][half * 8];
      bu.u[1] = *(const uint4*)&Us[nt * 16 + l16][half * 8 + 16];
      accg[nt] = wmma_bf16(af.v, bg.v, accg[nt]);
      accu[nt] = wmma_bf16(af.v, bu.v, accu[nt]);
    }
    __syncthreads();
  }
#pragma unroll
  for (int nt = 0; nt < 8; ++nt) {
#pragma unroll
    for (int r = 0; r < 8; ++r) {
      const int gm = m0 + wave * 16 + half * 8 + r;
      const int gn = n0 + nt * 16 + l16;
      const float g = accg[nt][r], u = accu[nt][r];
      const float silu = g / (1.f + __expf(-g));
      FFo[(size_t)gm * N + gn] = (bf16)(silu * u);
    }
  }
}

// ---------------------------------------------------------------------------
// Flash attention (causal).  Block = 128 query rows (8 waves x 16 rows),
// K/V streamed in 32-key LDS chunks (K async-to-LDS), online softmax in
// registers, QK^T and PV via v_wmma_f32_16x16x32_bf16.  qkv: [B,T,3,H,HD].
// ---------------------------------------------------------------------------
__global__ __launch_bounds__(256) void flash_attn(
    const bf16* __restrict__ qkv, bf16* __restrict__ out) {
  __shared__ __align__(16) bf16 Kt[32][128];      // [key][d]
  __shared__ __align__(16) bf16 Vt[128][32];      // [d][key] (transposed)
  __shared__ __align__(16) bf16 Pt[8][16][32];    // per-wave P tile [m][key]
  const int bh = blockIdx.y, b = bh / H_, h = bh % H_;
  const int qt = blockIdx.x;
  const int tid = threadIdx.x, wave = tid >> 5, lane = tid & 31;
  const int half = lane >> 4, l16 = lane & 15;
  const size_t rs = (size_t)3 * D_;
  const bf16* base = qkv + (size_t)b * T_ * rs + (size_t)h * HD_;

  FragBF qf[4];
  {
    const int qrow = qt * 128 + wave * 16 + l16;
    const bf16* qp = base + (size_t)qrow * rs;
#pragma unroll
    for (int f = 0; f < 4; ++f) {
      qf[f].u[0] = *(const uint4*)&qp[f * 32 + half * 8];
      qf[f].u[1] = *(const uint4*)&qp[f * 32 + half * 8 + 16];
    }
  }
  float mrow[8], lrow[8];
  v8f accO[8] = {};
#pragma unroll
  for (int r = 0; r < 8; ++r) { mrow[r] = -3.0e38f; lrow[r] = 0.f; }
  const float scale = 0.08838834764831845f;       // 1/sqrt(HD)
  const int jmax = qt * 4 + 3;

  for (int j = 0; j <= jmax; ++j) {
    const int kbeg = j * 32;
    for (int c = tid; c < 512; c += 256) {
      const int key = c >> 4, col = (c & 15) << 3;
      const bf16* kp = base + (size_t)(kbeg + key) * rs + D_;
      async_copy16(&kp[col], &Kt[key][col]);      // K rows: straight async DMA
      const bf16* vp = base + (size_t)(kbeg + key) * rs + 2 * D_;
      U4BF vv; vv.u = *(const uint4*)&vp[col];    // V needs transpose: via VGPRs
#pragma unroll
      for (int i = 0; i < 8; ++i) Vt[col + i][key] = vv.e[i];
    }
    async_wait();
    __syncthreads();

    // S = Q @ K^T  -> two 16x16 tiles; batch frag loads then WMMA chain
    FragBF kf[8];
#pragma unroll
    for (int f = 0; f < 4; ++f) {
      kf[f].u[0]     = *(const uint4*)&Kt[l16][f * 32 + half * 8];
      kf[f].u[1]     = *(const uint4*)&Kt[l16][f * 32 + half * 8 + 16];
      kf[4 + f].u[0] = *(const uint4*)&Kt[16 + l16][f * 32 + half * 8];
      kf[4 + f].u[1] = *(const uint4*)&Kt[16 + l16][f * 32 + half * 8 + 16];
    }
    v8f s0 = {}, s1 = {};
#pragma unroll
    for (int f = 0; f < 4; ++f) {
      s0 = wmma_bf16(qf[f].v, kf[f].v, s0);
      s1 = wmma_bf16(qf[f].v, kf[4 + f].v, s1);
    }

    // online softmax per row (row value lives across a 16-lane half)
#pragma unroll
    for (int r = 0; r < 8; ++r) {
      const int t = qt * 128 + wave * 16 + half * 8 + r;
      float a = s0[r] * scale, c2 = s1[r] * scale;
      if (kbeg + l16 > t)      a  = -3.0e38f;
      if (kbeg + 16 + l16 > t) c2 = -3.0e38f;
      float mx = fmaxf(a, c2);
#pragma unroll
      for (int msk = 8; msk >= 1; msk >>= 1) mx = fmaxf(mx, __shfl_xor(mx, msk));
      const float mnew = fmaxf(mrow[r], mx);
      const float p0 = __expf(a - mnew), p1 = __expf(c2 - mnew);
      float sum = p0 + p1;
#pragma unroll
      for (int msk = 8; msk >= 1; msk >>= 1) sum += __shfl_xor(sum, msk);
      const float corr = __expf(mrow[r] - mnew);
      lrow[r] = lrow[r] * corr + sum;
      mrow[r] = mnew;
#pragma unroll
      for (int nt = 0; nt < 8; ++nt) accO[nt][r] *= corr;
      const int mRow = half * 8 + r;
      Pt[wave][mRow][l16]      = (bf16)p0;
      Pt[wave][mRow][16 + l16] = (bf16)p1;
    }

    // O += P @ V
    FragBF pa, vb[8];
    pa.u[0] = *(const uint4*)&Pt[wave][l16][half * 8];
    pa.u[1] = *(const uint4*)&Pt[wave][l16][half * 8 + 16];
#pragma unroll
    for (int nt = 0; nt < 8; ++nt) {
      vb[nt].u[0] = *(const uint4*)&Vt[nt * 16 + l16][half * 8];
      vb[nt].u[1] = *(const uint4*)&Vt[nt * 16 + l16][half * 8 + 16];
    }
#pragma unroll
    for (int nt = 0; nt < 8; ++nt) accO[nt] = wmma_bf16(pa.v, vb[nt].v, accO[nt]);
    __syncthreads();
  }

  const int trow = qt * 128 + wave * 16 + half * 8;
#pragma unroll
  for (int nt = 0; nt < 8; ++nt) {
#pragma unroll
    for (int r = 0; r < 8; ++r) {
      const float o = accO[nt][r] / lrow[r];
      const int t = trow + r;
      const int d = nt * 16 + l16;
      out[((size_t)(b * T_ + t)) * D_ + h * HD_ + d] = (bf16)o;
    }
  }
}

// ---------------------------------------------------------------------------
// Driver
// ---------------------------------------------------------------------------
extern "C" void kernel_launch(void* const* d_in, const int* in_sizes, int n_in,
                              void* d_out, int out_size, void* d_ws, size_t ws_size,
                              hipStream_t stream) {
  (void)in_sizes; (void)n_in; (void)out_size; (void)ws_size;
  const float* x      = (const float*)d_in[0];
  const float* gamma  = (const float*)d_in[1];
  const float* beta   = (const float*)d_in[2];
  const float* qkv_w  = (const float*)d_in[3];
  const float* o_w    = (const float*)d_in[4];
  const float* gate_w = (const float*)d_in[5];
  const float* up_w   = (const float*)d_in[6];
  const float* down_w = (const float*)d_in[7];
  const float* n1w    = (const float*)d_in[8];
  const float* n2w    = (const float*)d_in[9];
  float* outp = (float*)d_out;

  char* p = (char*)d_ws;
  auto carve = [&](size_t bytes) -> void* {
    void* r = (void*)p; p += (bytes + 255) & ~(size_t)255; return r;
  };
  bf16* wqkv = (bf16*)carve((size_t)3 * D_ * D_ * 2);
  bf16* wo   = (bf16*)carve((size_t)D_ * D_ * 2);
  bf16* wg   = (bf16*)carve((size_t)FF_ * D_ * 2);
  bf16* wu   = (bf16*)carve((size_t)FF_ * D_ * 2);
  bf16* wd   = (bf16*)carve((size_t)D_ * FF_ * 2);
  bf16* hbuf = (bf16*)carve((size_t)M_ * D_ * 2);
  bf16* qkvb = (bf16*)carve((size_t)M_ * 3 * D_ * 2);
  bf16* attn = (bf16*)carve((size_t)M_ * D_ * 2);
  float* x1  = (float*)carve((size_t)M_ * D_ * 4);
  bf16* h2   = (bf16*)carve((size_t)M_ * D_ * 2);
  bf16* ff   = (bf16*)carve((size_t)M_ * FF_ * 2);

  cvt_f32_bf16<<<2048, 256, 0, stream>>>(qkv_w,  wqkv, (size_t)3 * D_ * D_);
  cvt_f32_bf16<<<2048, 256, 0, stream>>>(o_w,    wo,   (size_t)D_ * D_);
  cvt_f32_bf16<<<2048, 256, 0, stream>>>(gate_w, wg,   (size_t)FF_ * D_);
  cvt_f32_bf16<<<2048, 256, 0, stream>>>(up_w,   wu,   (size_t)FF_ * D_);
  cvt_f32_bf16<<<2048, 256, 0, stream>>>(down_w, wd,   (size_t)D_ * FF_);

  rmsnorm_affine<<<M_, 256, 0, stream>>>(x, n1w, gamma, beta, hbuf);
  gemm_bf16_tn<0><<<dim3(3 * D_ / 128, M_ / 128), 256, 0, stream>>>(
      hbuf, wqkv, (void*)qkvb, nullptr, M_, 3 * D_, D_);
  flash_attn<<<dim3(T_ / 128, B_ * H_), 256, 0, stream>>>(qkvb, attn);
  gemm_bf16_tn<1><<<dim3(D_ / 128, M_ / 128), 256, 0, stream>>>(
      attn, wo, (void*)x1, x, M_, D_, D_);

  rmsnorm_affine<<<M_, 256, 0, stream>>>(x1, n2w, gamma, beta, h2);
  gemm_gateup<<<dim3(FF_ / 128, M_ / 128), 256, 0, stream>>>(h2, wg, wu, ff, M_, FF_, D_);
  gemm_bf16_tn<1><<<dim3(D_ / 128, M_ / 128), 256, 0, stream>>>(
      ff, wd, (void*)outp, x1, M_, D_, FF_);
}
